// IMEXnet_59416577573185
// MI455X (gfx1250) — matrix operand order repaired and checked by reading, not compiled, not run on hardware
//
#include <hip/hip_runtime.h>
#include <hip/hip_bf16.h>
#include <math.h>

// ---------------------------------------------------------------------------
// IMEXnet forward for MI455X (gfx1250, wave32).
//  - conv3x3 / conv3x3T as implicit GEMM on v_wmma_f32_16x16x32_bf16,
//    A-operand double-buffered via global_load_async_to_lds_b128 (ASYNCcnt),
//    B-operand staged once per workgroup as raw bf16 input rows.
//  - training-mode BatchNorm via block reduction + float atomics
//  - (I + h L^T L)^-1 solve: closed-form filter t (9 sincos terms) stored
//    bit-reversed, full 128x128 complex FFT per workgroup in LDS
//    (DIF forward / DIT inverse => no bit-reversal passes).
// ---------------------------------------------------------------------------

typedef __bf16 bf16_t;
typedef __attribute__((ext_vector_type(16))) __bf16 v16bf;
typedef __attribute__((ext_vector_type(8)))  float  v8f;

#define NB     32
#define CH     64
#define HH     128
#define WW     128
#define KTOT   (CH * 9)        // 576
#define KSTEP  32
#define NKSTEP (KTOT / KSTEP)  // 18
#define H_STEP 0.1f
#define BN_EPS 1e-5f
#define FFT_PAD 130
#define ROW_PITCH 36           // 34 valid + pad (ushort units)

// ---------------------------------------------------------------------------
// Weight prep: zero-mean 3x3 kernels -> bf16 GEMM slabs.
// ---------------------------------------------------------------------------
__global__ void prep_weights_kernel(const float* __restrict__ K,
                                    bf16_t* __restrict__ Ac,
                                    bf16_t* __restrict__ At) {
  int id = blockIdx.x * blockDim.x + threadIdx.x;   // co*64 + ci
  if (id >= CH * CH) return;
  int co = id / CH, ci = id - co * CH;
  const float* kp = K + (size_t)id * 9;
  float v[9]; float m = 0.0f;
#pragma unroll
  for (int r = 0; r < 9; ++r) { v[r] = kp[r]; m += v[r]; }
  m *= (1.0f / 9.0f);
#pragma unroll
  for (int r = 0; r < 9; ++r) {
    float w = v[r] - m;
    int kh = r / 3, kw = r - kh * 3;
    Ac[co * KTOT + ci * 9 + r] = (bf16_t)w;
    At[ci * KTOT + co * 9 + (2 - kh) * 3 + (2 - kw)] = (bf16_t)w;
  }
}

// ---------------------------------------------------------------------------
// Closed-form FFT filter, stored bit-reversed in both axes.
// ---------------------------------------------------------------------------
__device__ __forceinline__ int brev7(int v) { return (int)(__brev((unsigned)v) >> 25); }

__global__ void build_filter_kernel(const float* __restrict__ L,
                                    float* __restrict__ tbr) {
  int c = blockIdx.x, u = blockIdx.y, v = threadIdx.x;
  const float* lp = L + (size_t)c * 9;
  float l[9]; float m = 0.0f;
#pragma unroll
  for (int r = 0; r < 9; ++r) { l[r] = lp[r]; m += l[r]; }
  m *= (1.0f / 9.0f);
  float re = 0.0f, im = 0.0f;
  const float w0 = -6.28318530717958647692f / 128.0f;
#pragma unroll
  for (int r = 0; r < 9; ++r) {
    int a = r / 3 - 1, b = r - (r / 3) * 3 - 1;
    float ang = w0 * (float)(u * a + v * b);
    float sn, cs; __sincosf(ang, &sn, &cs);
    float wv = l[r] - m;
    re += wv * cs; im += wv * sn;
  }
  float t = 1.0f / (H_STEP * (re * re + im * im) + 1.0f);
  tbr[((size_t)c * 128 + brev7(u)) * 128 + brev7(v)] = t;
}

// ---------------------------------------------------------------------------
// fp32 -> bf16 mirror (initial conv input).
// ---------------------------------------------------------------------------
__global__ void __launch_bounds__(256)
convert_kernel(const float* __restrict__ in, bf16_t* __restrict__ outb) {
  size_t total = (size_t)NB * CH * HH * WW;
  for (size_t i = (size_t)blockIdx.x * blockDim.x + threadIdx.x; i < total;
       i += (size_t)gridDim.x * blockDim.x)
    outb[i] = (bf16_t)in[i];
}

// ---------------------------------------------------------------------------
// Implicit-GEMM 3x3 conv, bf16 WMMA, f32 accumulate.
// 256 threads = 8 waves -> 64(co) x 32(pos) tile.
//  - B operand: 192 raw input rows (64ci x 3kh x 34) staged ONCE in LDS.
//    Fragment offsets are compile-time constants (no runtime div).
//  - A operand: 64x32 bf16 slab per K-step, double-buffered with
//    global_load_async_to_lds_b128 + s_wait_asynccnt (true overlap).
// addScale != 0: out = addsrc + addScale*acc (fused residual axpy for convT).
// ---------------------------------------------------------------------------
__global__ void __launch_bounds__(256)
conv_wmma_kernel(const bf16_t* __restrict__ xinb, const bf16_t* __restrict__ A,
                 const float* __restrict__ addsrc, float* __restrict__ out,
                 float addScale) {
  __shared__ unsigned short Abuf[2][CH * KSTEP];        // 2 x 4 KiB
  __shared__ unsigned short Rows[CH * 3 * ROW_PITCH];   // 13.5 KiB

  int tid  = threadIdx.x;
  int lane = tid & 31, wave = tid >> 5;
  int coBase  = (wave & 3) << 4;     // 0,16,32,48
  int posBase = (wave >> 2) << 4;    // 0,16
  int half = lane >> 4, m = lane & 15;

  int blk = blockIdx.x;              // NB*HH*(WW/32)
  int w0  = (blk & 3) << 5;
  int h   = (blk >> 2) & 127;
  int n   = blk >> 9;

  const unsigned short* xin = (const unsigned short*)xinb;

  // ---- issue async copy of A slab for K-step 0 (one b128 per lane) ----
  int co_l = tid >> 2, ch_l = tid & 3;  // 64 rows x 4 chunks of 16B
  unsigned long long gA0 = (unsigned long long)(const char*)A +
                           (unsigned)(co_l * KTOT * 2 + ch_l * 16);
  unsigned int lA = (unsigned int)(size_t)(&Abuf[0][0]) +
                    (unsigned)(co_l * KSTEP * 2 + ch_l * 16);
  asm volatile("global_load_async_to_lds_b128 %0, %1, off"
               :: "v"(lA), "v"(gA0) : "memory");

  // ---- stage input rows: ci in [0,64), kh in [0,3), 34 elems (padded) ----
  for (int i = tid; i < CH * 3 * ROW_PITCH; i += 256) {
    int row = i / ROW_PITCH, col = i - row * ROW_PITCH;
    if (col < 34) {
      int ci = row / 3, kh = row - ci * 3;
      int hh = h + kh - 1;
      int ww = w0 + col - 1;
      unsigned short v = 0;
      if ((unsigned)hh < (unsigned)HH && (unsigned)ww < (unsigned)WW)
        v = xin[(((size_t)n * CH + ci) << 14) + hh * WW + ww];
      Rows[i] = v;
    }
  }
  asm volatile("s_wait_asynccnt 0x0" ::: "memory");
  __syncthreads();

  union Frag { v16bf v; unsigned int u[8]; unsigned short s[16]; };
  v8f acc = {};
  int pos = posBase + m;

#pragma unroll
  for (int ks = 0; ks < NKSTEP; ++ks) {
    const int cur = ks & 1;
    // prefetch next A slab into the other buffer
    if (ks + 1 < NKSTEP) {
      unsigned long long g = (unsigned long long)(const char*)A +
          (unsigned)(co_l * KTOT * 2 + (ks + 1) * KSTEP * 2 + ch_l * 16);
      unsigned int l = (unsigned int)(size_t)(&Abuf[(ks + 1) & 1][0]) +
          (unsigned)(co_l * KSTEP * 2 + ch_l * 16);
      asm volatile("global_load_async_to_lds_b128 %0, %1, off"
                   :: "v"(l), "v"(g) : "memory");
    }

    Frag fa, fb;
    // A 16x32 bf16 layout (ISA 7.12.2): lane<16: VGPR g<4 K=2g..; g>=4 K=16+..;
    // lanes>=16 shift K by +8. Load as dwords (bf16 pairs).
    const unsigned int* Arow =
        (const unsigned int*)&Abuf[cur][(coBase + m) * KSTEP];
#pragma unroll
    for (int g = 0; g < 8; ++g) {
      int kl = ((g & 4) << 2) + (half << 3) + ((g & 3) << 1);
      fa.u[g] = Arow[kl >> 1];
    }
    // B 32x16 layout: element e -> K = (e&8)*2 + half*8 + (e&7), N = lane&15.
    // Row-buffer offset for K is a compile-time constant per (ks,e,half).
#pragma unroll
    for (int e = 0; e < 16; ++e) {
      const int kLo = ks * KSTEP + ((e & 8) << 1) + (e & 7);
      const int kHi = kLo + 8;
      const int oLo = ((kLo / 9) * 3 + (kLo % 9) / 3) * ROW_PITCH + (kLo % 9) % 3;
      const int oHi = ((kHi / 9) * 3 + (kHi % 9) / 3) * ROW_PITCH + (kHi % 9) % 3;
      int off = half ? oHi : oLo;
      fb.s[e] = Rows[off + pos];
    }
    acc = __builtin_amdgcn_wmma_f32_16x16x32_bf16(
        false, fa.v, false, fb.v, (short)0, acc, false, false);

    asm volatile("s_wait_asynccnt 0x0" ::: "memory");
    __syncthreads();
  }

  // C/D f32 16x16: VGPR r -> M = r + 8*half, N = lane&15
#pragma unroll
  for (int r = 0; r < 8; ++r) {
    int co = coBase + (half << 3) + r;
    size_t gi = (((size_t)n * CH + co) * HH + h) * WW + (w0 + pos);
    float v = acc[r];
    if (addScale != 0.0f) v = addsrc[gi] + addScale * v;
    out[gi] = v;
  }
}

// ---------------------------------------------------------------------------
// BatchNorm (training mode).
// ---------------------------------------------------------------------------
__global__ void zero_stats_kernel(float* stats) {
  if (threadIdx.x < 128) stats[threadIdx.x] = 0.0f;
}

__global__ void __launch_bounds__(256)
bn_stats_kernel(const float* __restrict__ buf, float* __restrict__ stats) {
  int c = blockIdx.x;
  int chunk = blockIdx.y;
  const int perBlk = (NB * HH * WW) / 32;    // 16384
  float s = 0.0f, ss = 0.0f;
  for (int i = threadIdx.x; i < perBlk; i += 256) {
    int e = chunk * perBlk + i;
    int n = e >> 14;
    int off = e & 16383;
    float v = buf[(((size_t)n * CH + c) << 14) + off];
    s += v; ss += v * v;
  }
  __shared__ float sm[256], sm2[256];
  sm[threadIdx.x] = s; sm2[threadIdx.x] = ss;
  __syncthreads();
  for (int st = 128; st > 0; st >>= 1) {
    if (threadIdx.x < st) {
      sm[threadIdx.x]  += sm[threadIdx.x + st];
      sm2[threadIdx.x] += sm2[threadIdx.x + st];
    }
    __syncthreads();
  }
  if (threadIdx.x == 0) {
    atomicAdd(&stats[c], sm[0]);
    atomicAdd(&stats[64 + c], sm2[0]);
  }
}

// Normalize (+optional ReLU); optionally emit bf16 mirror for the next conv.
__global__ void __launch_bounds__(256)
bn_apply_kernel(float* __restrict__ buf, const float* __restrict__ stats,
                const float* __restrict__ gamma, const float* __restrict__ beta,
                int doRelu, bf16_t* __restrict__ bfout) {
  const float inv = 1.0f / (float)(NB * HH * WW);
  size_t total = (size_t)NB * CH * HH * WW;
  for (size_t idx = (size_t)blockIdx.x * blockDim.x + threadIdx.x; idx < total;
       idx += (size_t)gridDim.x * blockDim.x) {
    int c = (int)((idx >> 14) & 63);
    float mean = stats[c] * inv;
    float var  = stats[64 + c] * inv - mean * mean;
    float sc   = gamma[c] * rsqrtf(var + BN_EPS);
    float v    = sc * (buf[idx] - mean) + beta[c];
    if (doRelu) v = fmaxf(v, 0.0f);
    buf[idx] = v;
    if (bfout) bfout[idx] = (bf16_t)v;
  }
}

// ---------------------------------------------------------------------------
// In-LDS 128x128 complex FFT solve per (n,c) image.
// ---------------------------------------------------------------------------
__device__ void fft_pass(float* re, float* im, int lineStride, int elemStride,
                         bool inverse) {
  int tid = threadIdx.x;
  const float TWO_PI = 6.28318530717958647692f;
  if (!inverse) {
    for (int len = 128; len >= 2; len >>= 1) {
      int half = len >> 1;
      for (int b = tid; b < 128 * 64; b += 256) {
        int line = b >> 6;
        int t = b & 63;
        int bl = t / half;
        int k  = t - bl * half;
        int i0 = bl * len + k, i1 = i0 + half;
        int a0 = line * lineStride + i0 * elemStride;
        int a1 = line * lineStride + i1 * elemStride;
        float ar = re[a0], ai = im[a0];
        float br = re[a1], bi = im[a1];
        float dr = ar - br, di = ai - bi;
        float ang = -TWO_PI * (float)k / (float)len;
        float sn, cs; __sincosf(ang, &sn, &cs);
        re[a0] = ar + br;            im[a0] = ai + bi;
        re[a1] = dr * cs - di * sn;  im[a1] = dr * sn + di * cs;
      }
      __syncthreads();
    }
  } else {
    for (int len = 2; len <= 128; len <<= 1) {
      int half = len >> 1;
      for (int b = tid; b < 128 * 64; b += 256) {
        int line = b >> 6;
        int t = b & 63;
        int bl = t / half;
        int k  = t - bl * half;
        int i0 = bl * len + k, i1 = i0 + half;
        int a0 = line * lineStride + i0 * elemStride;
        int a1 = line * lineStride + i1 * elemStride;
        float ang = TWO_PI * (float)k / (float)len;
        float sn, cs; __sincosf(ang, &sn, &cs);
        float br = re[a1], bi = im[a1];
        float tr = br * cs - bi * sn, ti = br * sn + bi * cs;
        float ar = re[a0], ai = im[a0];
        re[a0] = ar + tr; im[a0] = ai + ti;
        re[a1] = ar - tr; im[a1] = ai - ti;
      }
      __syncthreads();
    }
  }
}

__global__ void __launch_bounds__(256)
fft_solve_kernel(float* __restrict__ x, const float* __restrict__ tbr,
                 bf16_t* __restrict__ xbf) {
  extern __shared__ float lds[];
  float* re = lds;
  float* im = lds + 128 * FFT_PAD;
  int tid = threadIdx.x;
  int c = blockIdx.x & 63;
  size_t base = (size_t)blockIdx.x << 14;

  for (int i = tid; i < 16384; i += 256) {
    int row = i >> 7, col = i & 127;
    re[row * FFT_PAD + col] = x[base + i];
    im[row * FFT_PAD + col] = 0.0f;
  }
  __syncthreads();

  fft_pass(re, im, FFT_PAD, 1, false);  // rows forward (DIF)
  fft_pass(re, im, 1, FFT_PAD, false);  // cols forward (DIF)

  for (int i = tid; i < 16384; i += 256) {
    int row = i >> 7, col = i & 127;
    float f = tbr[((size_t)c << 14) + i];
    int a = row * FFT_PAD + col;
    re[a] *= f; im[a] *= f;
  }
  __syncthreads();

  fft_pass(re, im, 1, FFT_PAD, true);   // cols inverse (DIT)
  fft_pass(re, im, FFT_PAD, 1, true);   // rows inverse (DIT)

  const float scale = 1.0f / 16384.0f;
  for (int i = tid; i < 16384; i += 256) {
    int row = i >> 7, col = i & 127;
    float v = re[row * FFT_PAD + col] * scale;
    x[base + i] = v;
    xbf[base + i] = (bf16_t)v;   // bf16 mirror for next layer's conv
  }
}

// ---------------------------------------------------------------------------
extern "C" void kernel_launch(void* const* d_in, const int* in_sizes, int n_in,
                              void* d_out, int out_size, void* d_ws, size_t ws_size,
                              hipStream_t stream) {
  (void)in_sizes; (void)n_in; (void)out_size; (void)ws_size;
  const float* x_in  = (const float*)d_in[0];
  const float* K     = (const float*)d_in[1];
  const float* L     = (const float*)d_in[2];
  const float* gamma = (const float*)d_in[3];
  const float* beta  = (const float*)d_in[4];

  float* zbuf = (float*)d_out;   // d_out doubles as z scratch; overwritten at end

  char* ws = (char*)d_ws;
  size_t off = 0;
  const size_t nElem = (size_t)NB * CH * HH * WW;
  float*  x   = (float*)(ws + off);  off += nElem * sizeof(float);
  bf16_t* xbf = (bf16_t*)(ws + off); off += nElem * sizeof(bf16_t);
  bf16_t* zbf = (bf16_t*)(ws + off); off += nElem * sizeof(bf16_t);
  bf16_t* Ac  = (bf16_t*)(ws + off); off += (size_t)CH * KTOT * sizeof(bf16_t);
  bf16_t* At  = (bf16_t*)(ws + off); off += (size_t)CH * KTOT * sizeof(bf16_t);
  off = (off + 255) & ~(size_t)255;
  float* tbr   = (float*)(ws + off); off += (size_t)CH * HH * WW * sizeof(float);
  float* stats = (float*)(ws + off); off += 128 * sizeof(float);

  const size_t tensorBytes = nElem * sizeof(float);
  hipMemcpyAsync(x, x_in, tensorBytes, hipMemcpyDeviceToDevice, stream);
  convert_kernel<<<4096, 256, 0, stream>>>(x, xbf);

  const int convGrid = NB * HH * (WW / 32);                 // 16384 workgroups
  const size_t fftLds = (size_t)2 * 128 * FFT_PAD * sizeof(float);

  for (int j = 0; j < 4; ++j) {
    prep_weights_kernel<<<16, 256, 0, stream>>>(K + (size_t)j * CH * CH * 9, Ac, At);
    build_filter_kernel<<<dim3(CH, 128), 128, 0, stream>>>(L + (size_t)j * CH * 9, tbr);

    // z = conv3x3(x, Kproj)
    conv_wmma_kernel<<<convGrid, 256, 0, stream>>>(xbf, Ac, nullptr, zbuf, 0.0f);

    // z = relu(BN(z)), emit bf16 mirror for convT
    zero_stats_kernel<<<1, 128, 0, stream>>>(stats);
    bn_stats_kernel<<<dim3(CH, 32), 256, 0, stream>>>(zbuf, stats);
    bn_apply_kernel<<<4096, 256, 0, stream>>>(zbuf, stats, gamma + j * CH,
                                              beta + j * CH, 1, zbf);

    // x = x + h * conv3x3T(z, Kproj)   (fused axpy epilogue)
    conv_wmma_kernel<<<convGrid, 256, 0, stream>>>(zbf, At, x, x, H_STEP);

    // x = BN(x)  (fp32 only; FFT consumes fp32)
    zero_stats_kernel<<<1, 128, 0, stream>>>(stats);
    bn_stats_kernel<<<dim3(CH, 32), 256, 0, stream>>>(x, stats);
    bn_apply_kernel<<<4096, 256, 0, stream>>>(x, stats, gamma + j * CH,
                                              beta + j * CH, 0, nullptr);

    // x = (I + h L^T L)^-1 x  via in-LDS FFT; also emits next conv's bf16 input
    fft_solve_kernel<<<NB * CH, 256, fftLds, stream>>>(x, tbr, xbf);
  }

  hipMemcpyAsync(d_out, x, tensorBytes, hipMemcpyDeviceToDevice, stream);
}
